// CausalAttention_85976655331534
// MI455X (gfx1250) — matrix operand order repaired and checked by reading, not compiled, not run on hardware
//
#include <hip/hip_runtime.h>

// ---------------------------------------------------------------------------
// Causal attention (B=4, S=4096, Din=768, Dout=64), fp32 in/out.
// bf16 WMMA everywhere; flash-attention with transposed score tiles so the
// S^T accumulators ARE the A-operand of P@V (no LDS transpose).
// Register double-buffering so wmma chains overlap with operand loads.
// ---------------------------------------------------------------------------

typedef __bf16 bf16;
typedef __attribute__((ext_vector_type(16))) __bf16 bf16x16;
typedef __attribute__((ext_vector_type(8)))  __bf16 bf16x8;
typedef __attribute__((ext_vector_type(8)))  float  v8f;
typedef __attribute__((ext_vector_type(4)))  float  v4f;

#define DIN   768
#define DOUT  64
#define SEQ   4096
#define BATCH 4
#define NROWS (BATCH * SEQ)   // 16384
#define KSTEPS (DIN / 32)     // 24

// workspace layout (bytes)
#define WT_OFF  0                                  // 3 * (64 x 768) bf16  W^T
#define WT_SZ   (3 * DOUT * DIN * 2)               // 294912
#define QB_OFF  (WT_OFF + WT_SZ)                   // Q bf16 row-major (NROWS x 64)
#define QB_SZ   (NROWS * DOUT * 2)                 // 2 MiB
#define KB_OFF  (QB_OFF + QB_SZ)
#define VT_OFF  (KB_OFF + QB_SZ)                   // V^T bf16 (B, 64, SEQ)

__device__ __forceinline__ bf16x16 bcat(bf16x8 lo, bf16x8 hi) {
  return __builtin_shufflevector(lo, hi, 0,1,2,3,4,5,6,7,8,9,10,11,12,13,14,15);
}
// A operand (16-bit 16x32): lane(m, half) -> elems 0..7 = k half*8+w, 8..15 = k 16+half*8+w
__device__ __forceinline__ bf16x16 ldA(const bf16* rowp, int half) {
  const bf16* p = rowp + half * 8;
  return bcat(*(const bf16x8*)p, *(const bf16x8*)(p + 16));
}
// B operand (16-bit 32x16): lane(n, khalf) holds 16 contiguous k values
__device__ __forceinline__ bf16x16 ldB(const bf16* p) {
  return bcat(*(const bf16x8*)p, *(const bf16x8*)(p + 8));
}
__device__ __forceinline__ v8f wmma_bf16(bf16x16 a, bf16x16 b, v8f c) {
  return __builtin_amdgcn_wmma_f32_16x16x32_bf16(false, a, false, b, (short)0, c, false, false);
}
// 16 contiguous f32 -> bf16x16
__device__ __forceinline__ bf16x16 ldcvt16(const float* p) {
  v4f f0 = *(const v4f*)p,       f1 = *(const v4f*)(p + 4);
  v4f f2 = *(const v4f*)(p + 8), f3 = *(const v4f*)(p + 12);
  v8f lo = __builtin_shufflevector(f0, f1, 0,1,2,3,4,5,6,7);
  v8f hi = __builtin_shufflevector(f2, f3, 0,1,2,3,4,5,6,7);
  return bcat(__builtin_convertvector(lo, bf16x8),
              __builtin_convertvector(hi, bf16x8));
}
// x-row A operand: 8 f32 at p, 8 f32 at p+16 -> bf16x16
__device__ __forceinline__ bf16x16 ldcvtA(const float* p) {
  v4f a0 = *(const v4f*)p,        a1 = *(const v4f*)(p + 4);
  v4f b0 = *(const v4f*)(p + 16), b1 = *(const v4f*)(p + 20);
  v8f lo = __builtin_shufflevector(a0, a1, 0,1,2,3,4,5,6,7);
  v8f hi = __builtin_shufflevector(b0, b1, 0,1,2,3,4,5,6,7);
  return bcat(__builtin_convertvector(lo, bf16x8),
              __builtin_convertvector(hi, bf16x8));
}

// --------------------------- kernel 0: W -> W^T bf16 ------------------------
__global__ void prep_wt_kernel(const float* __restrict__ wq,
                               const float* __restrict__ wk,
                               const float* __restrict__ wv,
                               bf16* __restrict__ wt) {
  int idx = blockIdx.x * 256 + threadIdx.x;          // 3*768*64 = 147456 total
  if (idx >= 3 * DIN * DOUT) return;
  int p = idx / (DIN * DOUT);
  int r = idx - p * (DIN * DOUT);
  int n = r / DIN;          // out feature
  int k = r - n * DIN;      // in feature
  const float* w = (p == 0) ? wq : (p == 1) ? wk : wv;
  wt[(p * DOUT + n) * DIN + k] = (bf16)w[k * DOUT + n];
}

// ------------------- kernel 1: Q,K projection (out^T form) ------------------
__global__ void __launch_bounds__(256) proj_qk_kernel(
    const float* __restrict__ x, const float* __restrict__ bq,
    const float* __restrict__ bk, const bf16* __restrict__ wt,
    bf16* __restrict__ qb, bf16* __restrict__ kb) {
  int tid = threadIdx.x;
  int l = tid & 31, wave = tid >> 5;
  int gidx = blockIdx.x * 8 + wave;     // 0..2047
  int p  = gidx >> 10;                  // 0 = Q, 1 = K
  int st = gidx & 1023;
  int r0 = st * 16;
  const bf16*  Wt   = wt + p * DOUT * DIN;
  const float* bias = p ? bk : bq;
  bf16*        op   = p ? kb : qb;
  int lane15 = l & 15, half = l >> 4;

  const float* xrow = x + (r0 + lane15) * DIN + half * 8;   // B operand base
  const bf16*  wrow = Wt + lane15 * DIN;                    // A operand base

  v8f acc[4] = {v8f{}, v8f{}, v8f{}, v8f{}};
  // double-buffered operands
  bf16x16 xB = ldcvtA(xrow);
  bf16x16 w0 = ldA(wrow + 0 * 16 * DIN, half);
  bf16x16 w1 = ldA(wrow + 1 * 16 * DIN, half);
  bf16x16 w2 = ldA(wrow + 2 * 16 * DIN, half);
  bf16x16 w3 = ldA(wrow + 3 * 16 * DIN, half);
  for (int s = 0; s < KSTEPS - 1; ++s) {
    int kn = (s + 1) * 32;
    bf16x16 xBn = ldcvtA(xrow + kn);
    bf16x16 w0n = ldA(wrow + 0 * 16 * DIN + kn, half);
    bf16x16 w1n = ldA(wrow + 1 * 16 * DIN + kn, half);
    bf16x16 w2n = ldA(wrow + 2 * 16 * DIN + kn, half);
    bf16x16 w3n = ldA(wrow + 3 * 16 * DIN + kn, half);
    acc[0] = wmma_bf16(w0, xB, acc[0]);
    acc[1] = wmma_bf16(w1, xB, acc[1]);
    acc[2] = wmma_bf16(w2, xB, acc[2]);
    acc[3] = wmma_bf16(w3, xB, acc[3]);
    xB = xBn; w0 = w0n; w1 = w1n; w2 = w2n; w3 = w3n;
  }
  acc[0] = wmma_bf16(w0, xB, acc[0]);
  acc[1] = wmma_bf16(w1, xB, acc[1]);
  acc[2] = wmma_bf16(w2, xB, acc[2]);
  acc[3] = wmma_bf16(w3, xB, acc[3]);

#pragma unroll
  for (int mt = 0; mt < 4; ++mt) {
    const float* bp = bias + mt * 16 + 8 * half;
    bf16x8 o;
#pragma unroll
    for (int j = 0; j < 8; ++j) o[j] = (bf16)(acc[mt][j] + bp[j]);
    *(bf16x8*)(op + (r0 + lane15) * DOUT + mt * 16 + 8 * half) = o;
  }
}

// ---------------------- kernel 2: V projection (direct form) ----------------
__global__ void __launch_bounds__(256) proj_v_kernel(
    const float* __restrict__ x, const float* __restrict__ bv,
    const bf16* __restrict__ wtv, bf16* __restrict__ vt) {
  int tid = threadIdx.x;
  int l = tid & 31, wave = tid >> 5;
  int st = blockIdx.x * 8 + wave;       // 0..1023
  int r0 = st * 16;
  int lane15 = l & 15, half = l >> 4;

  const float* xrow = x + (r0 + lane15) * DIN + half * 16;  // A operand base
  const bf16*  wrow = wtv + lane15 * DIN + half * 16;       // B operand base

  v8f acc[4] = {v8f{}, v8f{}, v8f{}, v8f{}};
  bf16x16 xA = ldcvt16(xrow);
  bf16x16 w0 = ldB(wrow + 0 * 16 * DIN);
  bf16x16 w1 = ldB(wrow + 1 * 16 * DIN);
  bf16x16 w2 = ldB(wrow + 2 * 16 * DIN);
  bf16x16 w3 = ldB(wrow + 3 * 16 * DIN);
  for (int s = 0; s < KSTEPS - 1; ++s) {
    int kn = (s + 1) * 32;
    bf16x16 xAn = ldcvt16(xrow + kn);
    bf16x16 w0n = ldB(wrow + 0 * 16 * DIN + kn);
    bf16x16 w1n = ldB(wrow + 1 * 16 * DIN + kn);
    bf16x16 w2n = ldB(wrow + 2 * 16 * DIN + kn);
    bf16x16 w3n = ldB(wrow + 3 * 16 * DIN + kn);
    acc[0] = wmma_bf16(xA, w0, acc[0]);
    acc[1] = wmma_bf16(xA, w1, acc[1]);
    acc[2] = wmma_bf16(xA, w2, acc[2]);
    acc[3] = wmma_bf16(xA, w3, acc[3]);
    xA = xAn; w0 = w0n; w1 = w1n; w2 = w2n; w3 = w3n;
  }
  acc[0] = wmma_bf16(xA, w0, acc[0]);
  acc[1] = wmma_bf16(xA, w1, acc[1]);
  acc[2] = wmma_bf16(xA, w2, acc[2]);
  acc[3] = wmma_bf16(xA, w3, acc[3]);

  int b  = r0 >> 12;          // / SEQ
  int sl = r0 & (SEQ - 1);
#pragma unroll
  for (int nt = 0; nt < 4; ++nt) {
    int feat = nt * 16 + lane15;
    float bb = bv[feat];
    bf16x8 o;
#pragma unroll
    for (int j = 0; j < 8; ++j) o[j] = (bf16)(acc[nt][j] + bb);
    *(bf16x8*)(vt + (b * DOUT + feat) * SEQ + sl + 8 * half) = o;
  }
}

// ------------------------- kernel 3: flash attention ------------------------
__global__ void __launch_bounds__(256) attn_kernel(
    const bf16* __restrict__ qb, const bf16* __restrict__ kbm,
    const bf16* __restrict__ vt, float* __restrict__ out) {
  int tid = threadIdx.x;
  int l = tid & 31, wave = tid >> 5;
  int gtile = blockIdx.x * 8 + wave;    // 0..1023
  int b = gtile >> 8;                   // / 256 tiles per batch
  int t = gtile & 255;
  int q0 = t * 16;
  const bf16* Q = qb  + (size_t)b * SEQ * DOUT;
  const bf16* K = kbm + (size_t)b * SEQ * DOUT;
  const bf16* V = vt  + (size_t)b * DOUT * SEQ;
  int lane15 = l & 15, half = l >> 4;

  // Q as B operand (lane n = query, feats half*8+w and 16+half*8+w) — invariant
  const bf16* qp = Q + (q0 + lane15) * DOUT;
  bf16x16 qB0 = ldA(qp, half);
  bf16x16 qB1 = ldA(qp + 32, half);

  v8f O0 = {}, O1 = {}, O2 = {}, O3 = {};
  float m_run = -1e30f, l_run = 0.0f;
  const float scale = 0.125f;                 // 1/sqrt(64)
  const float L2E   = 1.4426950408889634f;
  int nb = (q0 + 16 + 31) >> 5;               // causal key blocks of 32

  const bf16* krb = K + lane15 * DOUT;
  // preload K block 0 A-operands
  bf16x16 a00 = ldA(krb, half);
  bf16x16 a01 = ldA(krb + 32, half);
  bf16x16 a10 = ldA(krb + 16 * DOUT, half);
  bf16x16 a11 = ldA(krb + 16 * DOUT + 32, half);

  for (int ib = 0; ib < nb; ++ib) {
    int kb = ib * 32;
    // S^T tiles: D(m=key_local, n=query)
    v8f s0 = {}, s1 = {};
    s0 = wmma_bf16(a00, qB0, s0);
    s0 = wmma_bf16(a01, qB1, s0);
    s1 = wmma_bf16(a10, qB0, s1);
    s1 = wmma_bf16(a11, qB1, s1);

    // issue V loads now so they hide under the softmax VALU below
    const bf16* vb = V + (size_t)lane15 * SEQ + kb;
    bf16x16 vB0 = ldA(vb, half);
    bf16x16 vB1 = ldA(vb + 16 * SEQ, half);
    bf16x16 vB2 = ldA(vb + 32 * SEQ, half);
    bf16x16 vB3 = ldA(vb + 48 * SEQ, half);

    // preload next K block while softmax math runs
    if (ib + 1 < nb) {
      const bf16* krn = krb + (kb + 32) * DOUT;
      a00 = ldA(krn, half);
      a01 = ldA(krn + 32, half);
      a10 = ldA(krn + 16 * DOUT, half);
      a11 = ldA(krn + 16 * DOUT + 32, half);
    }

    // causal mask + scale; per-query running softmax (query = lane15)
    int qg = q0 + lane15;
    float z[16];
    float mblk = -1e30f;
#pragma unroll
    for (int j = 0; j < 8; ++j) {
      int k0 = kb + half * 8 + j;
      float z0 = (k0      <= qg) ? s0[j] * scale : -1e30f;
      float z1 = (k0 + 16 <= qg) ? s1[j] * scale : -1e30f;
      z[j] = z0; z[j + 8] = z1;
      mblk = fmaxf(mblk, fmaxf(z0, z1));
    }
    mblk = fmaxf(mblk, __shfl_xor(mblk, 16, 32));      // combine key halves
    float m_new = fmaxf(m_run, mblk);
    float alpha = __builtin_amdgcn_exp2f((m_run - m_new) * L2E);
    float psum = 0.0f;
    bf16x16 aP;                                        // == A operand of P@V
#pragma unroll
    for (int i = 0; i < 16; ++i) {
      float pv = __builtin_amdgcn_exp2f((z[i] - m_new) * L2E);
      psum += pv;
      aP[i] = (bf16)pv;
    }
    psum += __shfl_xor(psum, 16, 32);
    l_run = l_run * alpha + psum;
    m_run = m_new;

    // rescale O (O lane: query = j + 8*half) with broadcast alpha
#pragma unroll
    for (int j = 0; j < 8; ++j) {
      float aj = __shfl(alpha, j + 8 * half, 32);
      O0[j] *= aj; O1[j] *= aj; O2[j] *= aj; O3[j] *= aj;
    }
    // P @ V
    O0 = wmma_bf16(aP, vB0, O0);
    O1 = wmma_bf16(aP, vB1, O1);
    O2 = wmma_bf16(aP, vB2, O2);
    O3 = wmma_bf16(aP, vB3, O3);
  }

  float inv = 1.0f / l_run;                   // lane holds stats for query lane15
  float* ob = out + ((size_t)b * SEQ + q0) * DOUT;
#pragma unroll
  for (int j = 0; j < 8; ++j) {
    float ij = __shfl(inv, j + 8 * half, 32);
    int r = (8 * half + j) * DOUT;
    ob[r +      lane15] = O0[j] * ij;
    ob[r + 16 + lane15] = O1[j] * ij;
    ob[r + 32 + lane15] = O2[j] * ij;
    ob[r + 48 + lane15] = O3[j] * ij;
  }
}

// ---------------------------------------------------------------------------
extern "C" void kernel_launch(void* const* d_in, const int* in_sizes, int n_in,
                              void* d_out, int out_size, void* d_ws, size_t ws_size,
                              hipStream_t stream) {
  const float* x  = (const float*)d_in[0];
  const float* wq = (const float*)d_in[1];
  const float* bq = (const float*)d_in[2];
  const float* wk = (const float*)d_in[3];
  const float* bk = (const float*)d_in[4];
  const float* wv = (const float*)d_in[5];
  const float* bv = (const float*)d_in[6];
  float* out = (float*)d_out;

  char* ws = (char*)d_ws;
  bf16* wt = (bf16*)(ws + WT_OFF);
  bf16* qb = (bf16*)(ws + QB_OFF);
  bf16* kb = (bf16*)(ws + KB_OFF);
  bf16* vt = (bf16*)(ws + VT_OFF);

  prep_wt_kernel<<<576, 256, 0, stream>>>(wq, wk, wv, wt);
  proj_qk_kernel<<<256, 256, 0, stream>>>(x, bq, bk, wt, qb, kb);
  proj_v_kernel <<<128, 256, 0, stream>>>(x, bv, wt + 2 * DOUT * DIN, vt);
  attn_kernel   <<<128, 256, 0, stream>>>(qb, kb, vt, out);
}
